// Quanv_77790447665907
// MI455X (gfx1250) — matrix operand order repaired and checked by reading, not compiled, not run on hardware
//
#include <hip/hip_runtime.h>

typedef __attribute__((ext_vector_type(8)))  __bf16 v8bf;
typedef __attribute__((ext_vector_type(16))) __bf16 v16bf;
typedef __attribute__((ext_vector_type(8)))  float  v8f;

#define BQ   32      // batch
#define CIN  64      // input channels
#define HW   64      // input H = W
#define OCH  128     // output channels
#define HO   62      // output H = W (VALID 3x3)
#define KTOT 576     // C * 9
#define KCH  64      // K staged per chunk (9 chunks)
#define NKT  18      // total 32-wide K tiles (576/32)
#define LDK  72      // padded LDS k-stride: 144B -> conflict-free, 16B aligned
#define WPLANE (OCH * KTOT)            // 73728 elems per bf16 weight plane
#define XELEMS (BQ * CIN * HW * HW)    // 8388608 x elements
#define WS_W   (2 * WPLANE * 2)        // 294912 B for weight planes
#define WS_X   (XELEMS * 4)            // 33554432 B for packed x

// fp32 -> (hi, lo) bf16 pair (RNE); hi+lo carries ~16 mantissa bits
__device__ __forceinline__ void split2(float v, __bf16 &h, __bf16 &l) {
  __bf16 hb = (__bf16)v;
  float hf = (float)hb;
  h = hb;
  l = (__bf16)(v - hf);
}

__device__ __forceinline__ v16bf load_frag(const __bf16* p0, const __bf16* p1) {
  v8bf a = *(const v8bf*)p0;      // ds_load_b128 / global_load_b128
  v8bf b = *(const v8bf*)p1;
  return __builtin_shufflevector(a, b, 0,1,2,3,4,5,6,7,8,9,10,11,12,13,14,15);
}

// ---------------------------------------------------------------------------
// Pre-pass 1: weights -> bf16 hi/lo planes in WMMA A-fragment order.
// slot = ((mt*NKT + kt)*2 + half)*32 + lane ; 8 contiguous bf16 per slot.
// ---------------------------------------------------------------------------
__global__ __launch_bounds__(256)
void prep_weights(const float* __restrict__ wgt,
                  __bf16* __restrict__ whi, __bf16* __restrict__ wlo) {
  int s = blockIdx.x * 256 + threadIdx.x;
  if (s >= (OCH / 16) * NKT * 2 * 32) return;   // 9216 slots
  int lane = s & 31;
  int half = (s >> 5) & 1;
  int kt   = (s >> 6) % NKT;
  int mt   = (s >> 6) / NKT;
  int oc = mt * 16 + (lane & 15);
  int kb = kt * 32 + half * 16 + (lane >> 4) * 8;
  #pragma unroll
  for (int j = 0; j < 8; ++j) {
    float v = wgt[oc * KTOT + kb + j];
    __bf16 h, l; split2(v, h, l);
    whi[s * 8 + j] = h;
    wlo[s * 8 + j] = l;
  }
}

// ---------------------------------------------------------------------------
// Pre-pass 2: x fp32 -> packed u32 (hi bf16 in [15:0], lo bf16 in [31:16]).
// Same byte count as fp32 -> no extra traffic in the main kernel.
// ---------------------------------------------------------------------------
__global__ __launch_bounds__(256)
void prep_x(const float* __restrict__ x, unsigned* __restrict__ xpk) {
  int idx = blockIdx.x * 1024 + threadIdx.x;
  #pragma unroll
  for (int j = 0; j < 4; ++j, idx += 256) {
    float v = x[idx];
    __bf16 h, l; split2(v, h, l);
    xpk[idx] = (unsigned)__builtin_bit_cast(unsigned short, h) |
               ((unsigned)__builtin_bit_cast(unsigned short, l) << 16);
  }
}

// ---------------------------------------------------------------------------
// Main kernel: one block per (b, ho); 8 waves cover 128 OC x 62 wo.
// im2col tile (64n x 64k) double-buffered in LDS; A frags straight from L2.
// Decode k->(c,kh,kw) is wave-uniform (readfirstlane) -> SALU pipe.
// ---------------------------------------------------------------------------
template <bool PACKED>
__global__ __launch_bounds__(256)
void quanv_wmma_bf16split(const float* __restrict__ x,
                          const unsigned* __restrict__ xpk,
                          const __bf16* __restrict__ whi,
                          const __bf16* __restrict__ wlo,
                          float* __restrict__ out) {
  __shared__ __align__(16) __bf16 sXh[2][64 * LDK];
  __shared__ __align__(16) __bf16 sXl[2][64 * LDK];

  const int t    = threadIdx.x;   // 0..255, 8 waves
  const int wv   = t >> 5;        // wave id -> OC rows [16*wv, 16*wv+16)
  const int lane = t & 31;
  const int g    = lane >> 4;     // lane half (WMMA operand layout)
  const int nl   = lane & 15;

  // staging coordinates: n fixed per thread (coalesced), k wave-uniform
  const int n    = t & 63;                                   // wo column
  const int qs   = __builtin_amdgcn_readfirstlane(t >> 6);   // uniform: SALU
  const int nLDK = n * LDK;

  const int b  = blockIdx.x / HO;
  const int ho = blockIdx.x % HO;
  const float*    xb  = x   + (b * CIN * HW + ho) * HW;
  const unsigned* xpb = xpk + (b * CIN * HW + ho) * HW;

  auto stageX = [&](int kc, int bufi) {
    const int ks = kc * KCH + qs;           // scalar k base
    __bf16* dh = sXh[bufi] + nLDK + qs;     // imm offset i*8B per store
    __bf16* dl = sXl[bufi] + nLDK + qs;
    #pragma unroll
    for (int i = 0; i < 16; ++i) {
      int k  = ks + i * 4;                  // unfold index (SALU)
      int c  = k / 9;                       // SALU magic-divide
      int r  = k - c * 9;
      int kh = r / 3;
      int kw = r - kh * 3;
      int soff = (c * HW + kh) * HW;        // scalar row offset
      int col = n + kw; col = col > HW - 1 ? HW - 1 : col;   // 2 VALU
      if constexpr (PACKED) {
        unsigned p = xpb[soff + col];       // saddr + voffset, b32
        dh[i * 4] = __builtin_bit_cast(__bf16, (unsigned short)p);
        dl[i * 4] = __builtin_bit_cast(__bf16, (unsigned short)(p >> 16));
      } else {
        float v = xb[soff + col];
        unsigned u = __builtin_bit_cast(unsigned, v);
        dh[i * 4] = __builtin_bit_cast(__bf16, (unsigned short)(u >> 16));
        float hf = __builtin_bit_cast(float, u & 0xffff0000u);
        dl[i * 4] = (__bf16)(v - hf);
      }
    }
  };

  v8f acc[4] = {};                 // 16(OC) x 64(wo), f32 accumulate
  stageX(0, 0);

  for (int kc = 0; kc < KTOT / KCH; ++kc) {
    __syncthreads();               // staged tile ready / other buffer free
    if (kc + 1 < KTOT / KCH)       // uniform: EXEC stays all-ones for WMMA
      stageX(kc + 1, (kc + 1) & 1);

    const __bf16* bufh = sXh[kc & 1];
    const __bf16* bufl = sXl[kc & 1];

    #pragma unroll
    for (int kt = 0; kt < 2; ++kt) {
      const int ktg = kc * 2 + kt;
      const __bf16* ph = whi + (((wv * NKT + ktg) * 2) * 32 + lane) * 8;
      const __bf16* pl = wlo + (((wv * NKT + ktg) * 2) * 32 + lane) * 8;
      v16bf ah = load_frag(ph, ph + 32 * 8);   // L2-resident A fragments
      v16bf al = load_frag(pl, pl + 32 * 8);
      #pragma unroll
      for (int nt = 0; nt < 4; ++nt) {
        const __bf16* xh = &bufh[(nt * 16 + nl) * LDK + kt * 32 + 8 * g];
        const __bf16* xl = &bufl[(nt * 16 + nl) * LDK + kt * 32 + 8 * g];
        v16bf bh = load_frag(xh, xh + 16);
        v16bf bl = load_frag(xl, xl + 16);
        acc[nt] = __builtin_amdgcn_wmma_f32_16x16x32_bf16(
            false, ah, false, bh, (short)0, acc[nt], false, false);
        acc[nt] = __builtin_amdgcn_wmma_f32_16x16x32_bf16(
            false, ah, false, bl, (short)0, acc[nt], false, false);
        acc[nt] = __builtin_amdgcn_wmma_f32_16x16x32_bf16(
            false, al, false, bh, (short)0, acc[nt], false, false);
      }
    }
  }

  // epilogue: replicate reference's layout bug: [oc][b][ho][wo]
  const int strideOC = BQ * HO * HO;            // 123008
  float* outb = out + b * (HO * HO) + ho * HO;
  #pragma unroll
  for (int nt = 0; nt < 4; ++nt) {
    int wo = nt * 16 + nl;
    if (wo < HO) {                              // drop pad columns 62,63
      #pragma unroll
      for (int v = 0; v < 8; ++v) {
        int oc = wv * 16 + 8 * g + v;           // D: VGPR v -> M = v + 8g
        outb[oc * strideOC + wo] = acc[nt][v];
      }
    }
  }
}

extern "C" void kernel_launch(void* const* d_in, const int* in_sizes, int n_in,
                              void* d_out, int out_size, void* d_ws, size_t ws_size,
                              hipStream_t stream) {
  const float* x = (const float*)d_in[0];   // (32, 64, 64, 64) fp32
  const float* w = (const float*)d_in[1];   // (128, 64, 9) fp32
  float* out = (float*)d_out;               // 128*32*62*62 fp32 (buggy layout)
  (void)in_sizes; (void)n_in; (void)out_size;

  __bf16*   whi = (__bf16*)d_ws;
  __bf16*   wlo = whi + WPLANE;
  unsigned* xpk = (unsigned*)((char*)d_ws + WS_W);

  prep_weights<<<dim3(36), dim3(256), 0, stream>>>(w, whi, wlo);

  if (ws_size >= (size_t)WS_W + WS_X) {        // fixed per run -> deterministic
    prep_x<<<dim3(XELEMS / 1024), dim3(256), 0, stream>>>(x, xpk);
    quanv_wmma_bf16split<true><<<dim3(BQ * HO), dim3(256), 0, stream>>>(
        x, xpk, whi, wlo, out);
  } else {
    quanv_wmma_bf16split<false><<<dim3(BQ * HO), dim3(256), 0, stream>>>(
        x, xpk, whi, wlo, out);
  }
}
